// LSA_42958262895152
// MI455X (gfx1250) — compile-verified, matrix operand order checked by reading, not run on hardware
//
#include <hip/hip_runtime.h>

typedef __attribute__((ext_vector_type(16))) _Float16 v16h;
typedef __attribute__((ext_vector_type(8)))  _Float16 v8h;
typedef __attribute__((ext_vector_type(8)))  float    v8f;

#define BATCH   8
#define TOK     1024
#define CDIM    1024
#define NHEAD   16
#define DHEAD   64
#define QKV_ELEMS (BATCH * NHEAD * TOK * DHEAD)   // 8,388,608 f16 elems per matrix

// ---------------------------------------------------------------------------
// Kernel 1: QKV projection  [8192,1024] x [1024,3072]
//   -> Q, K row-major f16 [B*h, tok, d];  V transposed f16 [B*h, d, tok]
// 256 threads = 8 waves; block tile 128(M) x 128(N); per-wave 32x64 output.
// W tile staged TRANSPOSED in LDS so B fragments are two ds_load_b128 each.
// ---------------------------------------------------------------------------
__global__ __launch_bounds__(256) void qkv_gemm(const float* __restrict__ x,
                                                const float* __restrict__ w,
                                                _Float16* __restrict__ qh,
                                                _Float16* __restrict__ kh,
                                                _Float16* __restrict__ vh) {
  __shared__ _Float16 As[128][40];   // [m][k]  row = 80B (16B multiple)
  __shared__ _Float16 Bt[128][40];   // [n][k]  transposed W tile

  const int tid  = threadIdx.x;
  const int lane = tid & 31;
  const int l16  = lane & 15;
  const int wid  = tid >> 5;
  const int wm   = wid & 3;          // 4 M-subtiles of 32
  const int wn   = wid >> 2;         // 2 N-subtiles of 64
  const int gm   = blockIdx.y * 128;
  const int gn   = blockIdx.x * 128;
  const int hb   = (lane < 16) ? 0 : 8;   // K-striping / row-group offset

  v8f acc[2][4] = {};

  const int arow = tid >> 1, ac0 = (tid & 1) * 16;  // A stage: 16 elems/thread
  const int bk   = tid >> 3, bn0 = (tid & 7) * 16;  // B stage: 16 elems/thread

  for (int k0 = 0; k0 < CDIM; k0 += 32) {
    __syncthreads();
    const float* xs = x + (size_t)(gm + arow) * CDIM + k0 + ac0;
#pragma unroll
    for (int j = 0; j < 16; ++j) As[arow][ac0 + j] = (_Float16)xs[j];
    const float* wp = w + (size_t)(k0 + bk) * 3072 + gn + bn0;
#pragma unroll
    for (int j = 0; j < 16; ++j) Bt[bn0 + j][bk] = (_Float16)wp[j];
    if (k0 + 32 < CDIM) {               // prefetch next k tile into caches
      __builtin_prefetch(xs + 32, 0, 1);
      __builtin_prefetch(wp + (size_t)32 * 3072, 0, 1);
    }
    __syncthreads();

    // A fragments (two 16-row subtiles)
    v16h a[2];
#pragma unroll
    for (int mi = 0; mi < 2; ++mi) {
      const _Float16* ar = &As[wm * 32 + mi * 16 + l16][0];
      v8h lo = *(const v8h*)(ar + hb);
      v8h ho = *(const v8h*)(ar + 16 + hb);
#pragma unroll
      for (int j = 0; j < 8; ++j) { a[mi][j] = lo[j]; a[mi][8 + j] = ho[j]; }
    }
#pragma unroll
    for (int f = 0; f < 4; ++f) {
      const _Float16* bp = &Bt[wn * 64 + f * 16 + l16][0];
      v8h lo = *(const v8h*)(bp + hb);
      v8h ho = *(const v8h*)(bp + 16 + hb);
      v16h bfr;
#pragma unroll
      for (int j = 0; j < 8; ++j) { bfr[j] = lo[j]; bfr[8 + j] = ho[j]; }
#pragma unroll
      for (int mi = 0; mi < 2; ++mi)
        acc[mi][f] = __builtin_amdgcn_wmma_f32_16x16x32_f16(
            false, a[mi], false, bfr, (short)0, acc[mi][f], false, false);
    }
  }

  // Scatter; per-wave 64-col block is head-aligned so which/head are uniform
  const int colbase = gn + wn * 64;
  const int which   = colbase >> 10;            // 0=q, 1=k, 2=v
  const int head    = (colbase & 1023) >> 6;
  if (which == 2) {
    // V transposed: [bh][dd][tok]; 8 consecutive toks -> one b128 store
#pragma unroll
    for (int mi = 0; mi < 2; ++mi) {
      const int grow0 = gm + wm * 32 + mi * 16 + hb;
      const int b = grow0 >> 10, tok0 = grow0 & 1023;
      _Float16* vb = vh + ((size_t)(b * NHEAD + head) * DHEAD) * TOK + tok0;
#pragma unroll
      for (int f = 0; f < 4; ++f) {
        const int dd = f * 16 + l16;
        v8h vv;
#pragma unroll
        for (int r = 0; r < 8; ++r) vv[r] = (_Float16)acc[mi][f][r];
        *(v8h*)(vb + (size_t)dd * TOK) = vv;
      }
    }
  } else {
    _Float16* outp = which ? kh : qh;           // row-major [bh][tok][dd]
#pragma unroll
    for (int mi = 0; mi < 2; ++mi) {
      const int grow0 = gm + wm * 32 + mi * 16 + hb;
#pragma unroll
      for (int f = 0; f < 4; ++f) {
        const int dd = f * 16 + l16;
#pragma unroll
        for (int r = 0; r < 8; ++r) {
          const int grow = grow0 + r;
          const int b = grow >> 10, tok = grow & 1023;
          outp[((size_t)((b * NHEAD + head) * TOK + tok)) * DHEAD + dd] =
              (_Float16)acc[mi][f][r];
        }
      }
    }
  }
}

// ---------------------------------------------------------------------------
// Kernel 2: causal attention, flash style. One wave = 16 query rows.
// V is consumed transposed -> B fragments are two global_load_b128 each.
// ---------------------------------------------------------------------------
__global__ __launch_bounds__(256) void attn(const _Float16* __restrict__ qh,
                                            const _Float16* __restrict__ kh,
                                            const _Float16* __restrict__ vh,
                                            const float* __restrict__ scale,
                                            float* __restrict__ out) {
  __shared__ _Float16 pbuf[8][16][40];  // per-wave P transpose bounce

  const int lane = threadIdx.x & 31;
  const int l16  = lane & 15;
  const int hb   = (lane < 16) ? 0 : 8;
  const int wid  = threadIdx.x >> 5;
  const int bh   = blockIdx.x >> 3;
  const int qg   = blockIdx.x & 7;
  const int b    = bh >> 4, h = bh & 15;
  const int q0   = (qg * 8 + wid) * 16;
  const float sc = scale[h];

  const _Float16* Q  = qh + (size_t)bh * TOK * DHEAD;
  const _Float16* K  = kh + (size_t)bh * TOK * DHEAD;
  const _Float16* Vt = vh + (size_t)bh * DHEAD * TOK;   // [d][tok]

  // Q fragments (A layout, 16x32 each over d = 0..31 and 32..63)
  v16h qa[2];
#pragma unroll
  for (int f = 0; f < 2; ++f) {
    const _Float16* qp = Q + (size_t)(q0 + l16) * DHEAD + f * 32;
    v8h lo = *(const v8h*)(qp + hb);
    v8h ho = *(const v8h*)(qp + 16 + hb);
#pragma unroll
    for (int j = 0; j < 8; ++j) { qa[f][j] = lo[j]; qa[f][8 + j] = ho[j]; }
  }

  v8f acc[4] = {};
  float rm[8], rl[8];
#pragma unroll
  for (int r = 0; r < 8; ++r) { rm[r] = -1e30f; rl[r] = 0.f; }

  for (int jc = 0; jc < q0 + 16; jc += 32) {   // wave-uniform causal bound
    // ---- S = Q @ K^T over 32 keys (two 16-key subtiles, K-dim = d = 64)
    v8f S[2];
#pragma unroll
    for (int s = 0; s < 2; ++s) {
      const int key = jc + s * 16 + l16;
      v8f sacc = {};
#pragma unroll
      for (int f = 0; f < 2; ++f) {
        const _Float16* kp = K + (size_t)key * DHEAD + f * 32;
        v8h lo = *(const v8h*)(kp + hb);
        v8h ho = *(const v8h*)(kp + 16 + hb);
        v16h kb;
#pragma unroll
        for (int j = 0; j < 8; ++j) { kb[j] = lo[j]; kb[8 + j] = ho[j]; }
        sacc = __builtin_amdgcn_wmma_f32_16x16x32_f16(
            false, qa[f], false, kb, (short)0, sacc, false, false);
      }
      S[s] = sacc;
    }
    // ---- online softmax, strict-lower mask (key < query)
#pragma unroll
    for (int r = 0; r < 8; ++r) {
      const int qrow = q0 + r + hb;
      const bool v0 = (jc + l16) < qrow;
      const bool v1 = (jc + 16 + l16) < qrow;
      float s0 = v0 ? S[0][r] * sc : -1e30f;
      float s1 = v1 ? S[1][r] * sc : -1e30f;
      float mx = fmaxf(s0, s1);
#pragma unroll
      for (int d = 1; d < 16; d <<= 1) mx = fmaxf(mx, __shfl_xor(mx, d, 32));
      const float mnew  = fmaxf(rm[r], mx);
      const float alpha = __expf(rm[r] - mnew);
      const float p0 = v0 ? __expf(s0 - mnew) : 0.f;
      const float p1 = v1 ? __expf(s1 - mnew) : 0.f;
      float ps = p0 + p1;
#pragma unroll
      for (int d = 1; d < 16; d <<= 1) ps += __shfl_xor(ps, d, 32);
      rl[r] = rl[r] * alpha + ps;
      rm[r] = mnew;
#pragma unroll
      for (int f = 0; f < 4; ++f) acc[f][r] *= alpha;
      pbuf[wid][r + hb][l16]      = (_Float16)p0;   // C-layout -> LDS
      pbuf[wid][r + hb][16 + l16] = (_Float16)p1;
    }
    asm volatile("s_wait_dscnt 0x0" ::: "memory");  // stores visible to reads
    // ---- reload P in A layout (16x32)
    v16h pa;
    {
      const _Float16* pr = &pbuf[wid][l16][0];
      v8h lo = *(const v8h*)(pr + hb);
      v8h ho = *(const v8h*)(pr + 16 + hb);
#pragma unroll
      for (int j = 0; j < 8; ++j) { pa[j] = lo[j]; pa[8 + j] = ho[j]; }
    }
    // ---- acc += P @ V  (V^T: keys contiguous per lane -> b128 loads)
#pragma unroll
    for (int f = 0; f < 4; ++f) {
      const int dd = f * 16 + l16;
      const _Float16* vp = Vt + (size_t)dd * TOK + jc;
      v8h lo = *(const v8h*)(vp + hb);
      v8h ho = *(const v8h*)(vp + 16 + hb);
      v16h vb;
#pragma unroll
      for (int j = 0; j < 8; ++j) { vb[j] = lo[j]; vb[8 + j] = ho[j]; }
      acc[f] = __builtin_amdgcn_wmma_f32_16x16x32_f16(
          false, pa, false, vb, (short)0, acc[f], false, false);
    }
  }

  // ---- normalize, zero fully-masked rows, store [B, N, h*d]
  float* op = out + ((size_t)b * TOK) * CDIM + (size_t)h * DHEAD;
#pragma unroll
  for (int f = 0; f < 4; ++f) {
    const int dd = f * 16 + l16;
#pragma unroll
    for (int r = 0; r < 8; ++r) {
      const int qrow = q0 + r + hb;
      const float lv = rl[r];
      op[(size_t)qrow * CDIM + dd] = (lv > 0.f) ? acc[f][r] / lv : 0.f;
    }
  }
}

extern "C" void kernel_launch(void* const* d_in, const int* in_sizes, int n_in,
                              void* d_out, int out_size, void* d_ws, size_t ws_size,
                              hipStream_t stream) {
  const float* x     = (const float*)d_in[0];
  const float* w     = (const float*)d_in[1];
  const float* scale = (const float*)d_in[2];
  _Float16* qhp = (_Float16*)d_ws;            // 16 MB
  _Float16* khp = qhp + (size_t)QKV_ELEMS;    // 16 MB
  _Float16* vhp = khp + (size_t)QKV_ELEMS;    // 16 MB (transposed layout)
  dim3 g1(3072 / 128, 8192 / 128);            // (24, 64) tiles
  qkv_gemm<<<g1, 256, 0, stream>>>(x, w, qhp, khp, vhp);
  attn<<<128 * 8, 256, 0, stream>>>(qhp, khp, vhp, scale, (float*)d_out);
}
